// Conv2d_69458211111150
// MI455X (gfx1250) — matrix operand order repaired and checked by reading, not compiled, not run on hardware
//
#include <hip/hip_runtime.h>

typedef __attribute__((ext_vector_type(16))) __bf16 v16bf;
typedef __attribute__((ext_vector_type(8)))  float  v8f;

#define CH   256
#define OCH  256
#define IH   224
#define IW   224
#define HW   (IH * IW)

// pre-split weights: 9 kpos x 8 c-chunks x 256 o x 2 halves x 16 elems
#define ATOT (9 * 8 * 256 * 2 * 16)            // 589824 bf16 per array
// pre-transposed x with 1-px zero halo: [hy+1][cch][wx+1][k32], k32 in B-frag order
#define XROWS 226
#define XCOLS 226
#define XTOT  (XROWS * 8 * XCOLS * 32)         // 13,075,456 bf16 per array

#define WMMA_BF16(A, B, Cacc) \
    __builtin_amdgcn_wmma_f32_16x16x32_bf16(false, (A), false, (B), (short)0, (Cacc), false, false)

// ---------------------------------------------------------------------------
// Prep 1: weights -> bf16 hi/lo in WMMA A-fragment lane order.
//   t = (((kpos*8 + cch)*256 + o)*2 + s)*16 + e,  k = s*16 + e
// ---------------------------------------------------------------------------
__global__ __launch_bounds__(256) void prep_weights(
    const float* __restrict__ wgt, __bf16* __restrict__ aHi, __bf16* __restrict__ aLo)
{
    const int t    = blockIdx.x * 256 + threadIdx.x;
    const int e    = t & 15;
    const int s    = (t >> 4) & 1;
    const int o    = (t >> 5) & 255;
    const int cch  = (t >> 13) & 7;
    const int kpos = t >> 16;
    const int k    = s * 16 + e;

    const float  v  = wgt[((long)o * CH + cch * 32 + k) * 9 + kpos];
    const __bf16 hi = (__bf16)v;
    aHi[t] = hi;
    aLo[t] = (__bf16)(v - (float)hi);
}

// ---------------------------------------------------------------------------
// Prep 2: x -> bf16 hi/lo, transposed to B-fragment order with zero halo.
//   xT[(hyp*8 + cch)*226 + wxp][k] , k = 0..31 linear channel-in-chunk
//   (B frag: lane s*16+col reads k = s*16 + e contiguously)
// One block per (hyp, wxp); 256 threads = 8 cch x 32 k.
// ---------------------------------------------------------------------------
__global__ __launch_bounds__(256) void prep_x(
    const float* __restrict__ x, __bf16* __restrict__ xHi, __bf16* __restrict__ xLo)
{
    const int hyp = blockIdx.x / XCOLS;        // 0..225
    const int wxp = blockIdx.x % XCOLS;        // 0..225
    const int cch = threadIdx.x >> 5;          // 0..7
    const int k   = threadIdx.x & 31;          // 0..31

    const int hy = hyp - 1, wx = wxp - 1;
    float v = 0.0f;
    if ((unsigned)hy < (unsigned)IH && (unsigned)wx < (unsigned)IW)
        v = x[(long)(cch * 32 + k) * HW + (long)hy * IW + wx];

    const long t  = ((long)(hyp * 8 + cch) * XCOLS + wxp) * 32 + k;
    const __bf16 hi = (__bf16)v;
    xHi[t] = hi;
    xLo[t] = (__bf16)(v - (float)hi);
}

// ---------------------------------------------------------------------------
// Main: implicit GEMM conv (9 shifted GEMMs), bf16x3 on the WMMA pipe.
// Block = 256 thr (8 waves) -> 256 o-ch x 32 px; wave = 32 o x 32 px.
// All fragments are contiguous 32B global loads (L2-resident: 2.25 + 52 MB
// << 192 MB L2). No LDS, no barriers, no predication in the hot loop.
// 12 WMMAs per K-chunk, round-robin over 4 accumulators (RAW distance 4).
// ---------------------------------------------------------------------------
__global__ __launch_bounds__(256) void conv2d_wmma_bf16x3(
    const __bf16* __restrict__ aHi, const __bf16* __restrict__ aLo,
    const __bf16* __restrict__ xHi, const __bf16* __restrict__ xLo,
    float* __restrict__ out)
{
    const int lane = threadIdx.x & 31;
    const int wave = threadIdx.x >> 5;

    const int h     = blockIdx.x / 7;          // output row
    const int w0    = (blockIdx.x % 7) * 32;   // first output col
    const int obase = wave * 32;

    const int s    = lane >> 4;                // lane half -> K sub-block
    const int arow = obase + (lane & 15);
    const int bcol = lane & 15;

    const long aoff = (long)arow * 32 + s * 16;   // within 256x32 chunk block
    const long boff = (long)bcol * 32 + s * 16;   // within row-of-cols block

    v8f acc00 = {}, acc01 = {}, acc10 = {}, acc11 = {};

    for (int kh = 0; kh < 3; ++kh) {
        for (int kw = 0; kw < 3; ++kw) {
            const int kpos = kh * 3 + kw;
            for (int cch = 0; cch < 8; ++cch) {
                const long abase = (long)(kpos * 8 + cch) * (256 * 32) + aoff;
                // padded col = w0 + kw + px; padded row = h + kh  (halo = +1)
                const long bbase =
                    ((long)((h + kh) * 8 + cch) * XCOLS + (w0 + kw)) * 32 + boff;

                const v16bf ahi0 = *(const v16bf*)(aHi + abase);
                const v16bf alo0 = *(const v16bf*)(aLo + abase);
                const v16bf ahi1 = *(const v16bf*)(aHi + abase + 16 * 32);
                const v16bf alo1 = *(const v16bf*)(aLo + abase + 16 * 32);

                const v16bf bhi0 = *(const v16bf*)(xHi + bbase);
                const v16bf blo0 = *(const v16bf*)(xLo + bbase);
                const v16bf bhi1 = *(const v16bf*)(xHi + bbase + 16 * 32);
                const v16bf blo1 = *(const v16bf*)(xLo + bbase + 16 * 32);

                acc00 = WMMA_BF16(ahi0, bhi0, acc00);
                acc01 = WMMA_BF16(ahi0, bhi1, acc01);
                acc10 = WMMA_BF16(ahi1, bhi0, acc10);
                acc11 = WMMA_BF16(ahi1, bhi1, acc11);

                acc00 = WMMA_BF16(ahi0, blo0, acc00);
                acc01 = WMMA_BF16(ahi0, blo1, acc01);
                acc10 = WMMA_BF16(ahi1, blo0, acc10);
                acc11 = WMMA_BF16(ahi1, blo1, acc11);

                acc00 = WMMA_BF16(alo0, bhi0, acc00);
                acc01 = WMMA_BF16(alo0, bhi1, acc01);
                acc10 = WMMA_BF16(alo1, bhi0, acc10);
                acc11 = WMMA_BF16(alo1, bhi1, acc11);
            }
        }
    }

    // store: VGPR r holds M = r + 8*s; cols = w0 + g*16 + bcol
    const int  orow0 = obase + s * 8;
    const long pixb  = (long)h * IW + w0 + bcol;
    #pragma unroll
    for (int r = 0; r < 8; ++r) {
        out[(long)(orow0 + r)      * HW + pixb     ] = acc00[r];
        out[(long)(orow0 + r)      * HW + pixb + 16] = acc01[r];
        out[(long)(orow0 + 16 + r) * HW + pixb     ] = acc10[r];
        out[(long)(orow0 + 16 + r) * HW + pixb + 16] = acc11[r];
    }
}

extern "C" void kernel_launch(void* const* d_in, const int* in_sizes, int n_in,
                              void* d_out, int out_size, void* d_ws, size_t ws_size,
                              hipStream_t stream) {
    const float* x = (const float*)d_in[0];   // [256,224,224]
    const float* w = (const float*)d_in[1];   // [256,256,3,3]
    float* out = (float*)d_out;               // [256,224,224]

    __bf16* aHi = (__bf16*)d_ws;              // 589824
    __bf16* aLo = aHi + ATOT;                 // 589824
    __bf16* xHi = aLo + ATOT;                 // 13,075,456
    __bf16* xLo = xHi + XTOT;                 // 13,075,456  (~54.7 MB total)

    prep_weights<<<ATOT / 256, 256, 0, stream>>>(w, aHi, aLo);
    prep_x<<<XROWS * XCOLS, 256, 0, stream>>>(x, xHi, xLo);

    conv2d_wmma_bf16x3<<<7 * 224, 256, 0, stream>>>(aHi, aLo, xHi, xLo, out);
}
